// GAT_42949672960529
// MI455X (gfx1250) — matrix operand order repaired and checked by reading, not compiled, not run on hardware
//
#include <hip/hip_runtime.h>
#include <hip/hip_bf16.h>

// ---------- types ----------
typedef __attribute__((ext_vector_type(16))) __bf16   v16bf;
typedef __attribute__((ext_vector_type(8)))  float    v8f;
typedef __attribute__((ext_vector_type(4)))  unsigned u32x4;
typedef __attribute__((ext_vector_type(8)))  int      i32x8;
typedef __attribute__((ext_vector_type(4)))  int      i32x4;

#define IN_C   128
#define KDIM   128      // inner dim of every GEMM in this net
#define HIDC   32
#define HEADS  4
#define OUT_C  8
#define NEG_SLOPE 0.2f

// ---------- helpers ----------
__device__ __forceinline__ unsigned f2key(float f) {
    unsigned u = __float_as_uint(f);
    return (u & 0x80000000u) ? ~u : (u | 0x80000000u);
}
__device__ __forceinline__ float key2f(unsigned k) {
    unsigned u = (k & 0x80000000u) ? (k & 0x7FFFFFFFu) : ~k;
    return __uint_as_float(u);
}
__device__ __forceinline__ float lrelu(float v) {
    return v >= 0.f ? v : NEG_SLOPE * v;
}

// TDM: 1-D contiguous copy global -> LDS, 8-byte elements.
// lds_off / nbytes must be 8B aligned; nbytes/8 must fit in 16 bits.
__device__ __forceinline__ void tdm_copy_to_lds(const void* gsrc, unsigned lds_off,
                                                unsigned nbytes) {
    unsigned long long ga = (unsigned long long)(size_t)gsrc;
    unsigned elems = nbytes >> 3;                 // 8-byte units
    u32x4 g0;
    g0[0] = 1u;                                   // count=1 (user descriptor)
    g0[1] = lds_off;                              // lds_addr (bytes)
    g0[2] = (unsigned)(ga & 0xFFFFFFFFu);         // global_addr[31:0]
    g0[3] = (unsigned)((ga >> 32) & 0x1FFFFFFu)   // global_addr[56:32]
          | (2u << 30);                           // type = 2 ("image")
    i32x8 g1;
    g1[0] = (int)(3u << 16);                      // wg_mask=0 | data_size=3 (8B)
    g1[1] = (int)((elems & 0xFFFFu) << 16);       // tensor_dim0[15:0]
    g1[2] = (int)(((elems >> 16) & 0xFFFFu)       // tensor_dim0[31:16]
          | (1u << 16));                          // tensor_dim1 = 1
    g1[3] = (int)((elems & 0xFFFFu) << 16);       // tile_dim0 = elems
    g1[4] = 1;                                    // tile_dim1 = 1
    g1[5] = (int)elems;                           // tensor_dim0_stride[31:0]
    g1[6] = 0;
    g1[7] = 0;
    i32x4 z4 = {0, 0, 0, 0};
#if defined(__clang_major__) && (__clang_major__ >= 23)
    i32x8 z8 = {0, 0, 0, 0, 0, 0, 0, 0};
    __builtin_amdgcn_tensor_load_to_lds(g0, g1, z4, z4, z8, 0);
#else
    __builtin_amdgcn_tensor_load_to_lds(g0, g1, z4, z4, 0);
#endif
}

// ---------- zero fill ----------
__global__ void zero_f32_k(float* __restrict__ p, int n) {
    int t = blockIdx.x * blockDim.x + threadIdx.x;
    if (t < n) p[t] = 0.f;
}

// ---------- fp32 -> bf16 conversions (done once per GEMM) ----------
__global__ void cvt_bf16_k(const float* __restrict__ src, __bf16* __restrict__ dst, int n) {
    int t = blockIdx.x * blockDim.x + threadIdx.x;
    if (t < n) dst[t] = (__bf16)src[t];
}
// Wt[n][k] = (bf16) W[k][n]   (W is [K, MC] row-major)
__global__ void cvt_wt_k(const float* __restrict__ W, __bf16* __restrict__ Wt, int MC) {
    int i = blockIdx.x * blockDim.x + threadIdx.x;
    if (i >= MC * KDIM) return;
    int n = i / KDIM, k = i - n * KDIM;
    Wt[i] = (__bf16)W[k * MC + n];
}

// ---------- dense GEMM via WMMA bf16, TDM-staged LDS ----------
// OUT[N, MC] = X[N, 128] @ W[128, MC], X/Wt pre-converted to bf16, Wt transposed.
// grid.x = ceil(N/64); block = 32*(MC/16) threads; each wave owns one 16-col tile
// and iterates 4 row-tiles, reusing its B operand registers across them.
// LDS: Wt bf16 [MC][128] at offset 0, As bf16 [64][128] after it.
__global__ void gemm_bf16_wmma_k(const __bf16* __restrict__ Xbf,
                                 const __bf16* __restrict__ Wtbf,
                                 float* __restrict__ OUT,
                                 int N, int MC) {
    extern __shared__ __bf16 smem[];
    __bf16* Wt = smem;                 // [MC][KDIM]
    __bf16* As = smem + MC * KDIM;     // [64][KDIM]

    const int row0 = blockIdx.x * 64;

    // ---- TDM staging (wave 0 issues, everyone waits at the barrier) ----
    if (threadIdx.x < 32) {
        const unsigned wbytes = (unsigned)(MC * KDIM * sizeof(__bf16));
        tdm_copy_to_lds(Wtbf, 0u, wbytes);
        int rows = N - row0; if (rows > 64) rows = 64;
        if (rows > 0) {
            tdm_copy_to_lds(Xbf + (size_t)row0 * KDIM, wbytes,
                            (unsigned)(rows * KDIM * sizeof(__bf16)));
        }
        __builtin_amdgcn_s_wait_tensorcnt(0);
    }
    __syncthreads();

    const int wave = threadIdx.x >> 5;     // column tile id
    const int lane = threadIdx.x & 31;
    const int c0   = wave * 16;
    const int m16  = lane & 15;
    const int hi   = lane >> 4;

    v8f acc[4] = {v8f{}, v8f{}, v8f{}, v8f{}};

    #pragma unroll
    for (int kt = 0; kt < KDIM; kt += 32) {
        // B 32x16 bf16 layout: lane col n = c0+m16; K base = kt + hi*8,
        // VGPR j=0..3 -> K = base+2j,{+1}; j=4..7 -> K = base+16+2(j-4),{+1}
        v16bf b;
        const int bkb = kt + hi * 8;
        const __bf16* br = Wt + (c0 + m16) * KDIM;
        #pragma unroll
        for (int j = 0; j < 4; ++j) {
            b[2 * j]     = br[bkb + 2 * j];
            b[2 * j + 1] = br[bkb + 2 * j + 1];
            b[2 * j + 8] = br[bkb + 16 + 2 * j];
            b[2 * j + 9] = br[bkb + 16 + 2 * j + 1];
        }
        // A 16x32 bf16 layout: lane row = m16; K base = kt + hi*8,
        // VGPR j=0..3 -> K = base+2j,{+1}; j=4..7 -> K = base+16+2(j-4),{+1}
        const int akb = kt + hi * 8;
        #pragma unroll
        for (int rt = 0; rt < 4; ++rt) {
            const __bf16* ar = As + (rt * 16 + m16) * KDIM;
            v16bf a;
            #pragma unroll
            for (int j = 0; j < 4; ++j) {
                a[2 * j]     = ar[akb + 2 * j];
                a[2 * j + 1] = ar[akb + 2 * j + 1];
                a[2 * j + 8] = ar[akb + 16 + 2 * j];
                a[2 * j + 9] = ar[akb + 16 + 2 * j + 1];
            }
            acc[rt] = __builtin_amdgcn_wmma_f32_16x16x32_bf16(
                false, a, false, b, (short)0, acc[rt], false, false);
        }
    }

    // C/D layout: VGPR r -> row M = r + hi*8, col N = m16
    #pragma unroll
    for (int rt = 0; rt < 4; ++rt) {
        #pragma unroll
        for (int r = 0; r < 8; ++r) {
            int m = row0 + rt * 16 + r + hi * 8;
            if (m < N) OUT[(size_t)m * MC + c0 + m16] = acc[rt][r];
        }
    }
}

// ---------- attention coefficients ----------
__global__ void attn_coef_k(const float* __restrict__ htmp,
                            const float* __restrict__ a_src,
                            const float* __restrict__ a_dst,
                            float* __restrict__ als, float* __restrict__ ald,
                            int N, int H, int C) {
    int t = blockIdx.x * blockDim.x + threadIdx.x;
    if (t >= N * H) return;
    int n = t / H, h = t - n * H;
    const float* hp = htmp + (size_t)(n * H + h) * C;
    float s = 0.f, d = 0.f;
    for (int c = 0; c < C; ++c) {
        float v = hp[c];
        s += v * a_src[h * C + c];
        d += v * a_dst[h * C + c];
    }
    als[t] = s;
    ald[t] = d;
}

// ---------- edge pass 1: segment max (monotonic uint encoding) ----------
__global__ void edge_max_k(const int* __restrict__ ei, int E0, int N, int H,
                           const float* __restrict__ als, const float* __restrict__ ald,
                           unsigned* __restrict__ maxkey) {
    int e = blockIdx.x * blockDim.x + threadIdx.x;
    int Etot = E0 + N;
    if (e >= Etot) return;
    int s, d;
    if (e < E0) { s = ei[e]; d = ei[E0 + e]; } else { s = d = e - E0; }
    for (int h = 0; h < H; ++h) {
        float v = lrelu(als[s * H + h] + ald[d * H + h]);
        atomicMax(&maxkey[d * H + h], f2key(v));
    }
}

// ---------- edge pass 2: denom = segment_sum(exp(e - m)) ----------
__global__ void edge_sum_k(const int* __restrict__ ei, int E0, int N, int H,
                           const float* __restrict__ als, const float* __restrict__ ald,
                           const unsigned* __restrict__ maxkey,
                           float* __restrict__ denom) {
    int e = blockIdx.x * blockDim.x + threadIdx.x;
    int Etot = E0 + N;
    if (e >= Etot) return;
    int s, d;
    if (e < E0) { s = ei[e]; d = ei[E0 + e]; } else { s = d = e - E0; }
    for (int h = 0; h < H; ++h) {
        float v = lrelu(als[s * H + h] + ald[d * H + h]);
        float m = key2f(maxkey[d * H + h]);
        if (!isfinite(m)) m = 0.f;
        atomicAdd(&denom[d * H + h], expf(v - m));
    }
}

// ---------- edge pass 3: weighted scatter of source features ----------
__global__ void edge_scatter_k(const int* __restrict__ ei, int E0, int N, int H, int C,
                               const float* __restrict__ als, const float* __restrict__ ald,
                               const unsigned* __restrict__ maxkey,
                               const float* __restrict__ denom,
                               const float* __restrict__ htmp,
                               float* __restrict__ acc) {
    int t = blockIdx.x * blockDim.x + threadIdx.x;
    int Etot = E0 + N;
    if (t >= Etot * H) return;
    int e = t / H, h = t - e * H;
    int s, d;
    if (e < E0) { s = ei[e]; d = ei[E0 + e]; } else { s = d = e - E0; }
    const float* hp = htmp + (size_t)(s * H + h) * C;
    __builtin_prefetch(hp, 0, 0);       // global_prefetch_b8
    float v = lrelu(als[s * H + h] + ald[d * H + h]);
    float m = key2f(maxkey[d * H + h]);
    if (!isfinite(m)) m = 0.f;
    float alpha = expf(v - m) / fmaxf(denom[d * H + h], 1e-16f);
    float* op = acc + (size_t)(d * H + h) * C;
    const float4* hp4 = reinterpret_cast<const float4*>(hp);
    #pragma unroll 4
    for (int q = 0; q < C / 4; ++q) {
        float4 v4 = hp4[q];             // global_load_b128 gather
        atomicAdd(&op[4 * q + 0], v4.x * alpha);
        atomicAdd(&op[4 * q + 1], v4.y * alpha);
        atomicAdd(&op[4 * q + 2], v4.z * alpha);
        atomicAdd(&op[4 * q + 3], v4.w * alpha);
    }
}

// ---------- bias + ELU ----------
__global__ void bias_elu_k(const float* __restrict__ acc, const float* __restrict__ b,
                           float* __restrict__ out, int N, int HC) {
    int t = blockIdx.x * blockDim.x + threadIdx.x;
    if (t >= N * HC) return;
    int j = t % HC;
    float v = acc[t] + b[j];
    out[t] = v > 0.f ? v : (expf(v) - 1.f);
}

// ---------- global mean pool accumulation ----------
__global__ void pool_accum_k(const float* __restrict__ h2, const int* __restrict__ batch,
                             float* __restrict__ sums, float* __restrict__ cnts, int N) {
    int t = blockIdx.x * blockDim.x + threadIdx.x;
    if (t >= N * HIDC) return;
    int n = t >> 5, c = t & 31;
    int g = batch[n];
    atomicAdd(&sums[g * HIDC + c], h2[t]);
    if (c == 0) atomicAdd(&cnts[g], 1.0f);
}

// ---------- final MLP on pooled graph vectors ----------
__global__ void mlp_k(const float* __restrict__ sums, const float* __restrict__ cnts,
                      const float* __restrict__ w1, const float* __restrict__ b1,
                      const float* __restrict__ w2, const float* __restrict__ b2,
                      float* __restrict__ out, int G) {
    int g = blockIdx.x * blockDim.x + threadIdx.x;
    if (g >= G) return;
    float gv[HIDC];
    float cnt = fmaxf(cnts[g], 1.0f);
    for (int c = 0; c < HIDC; ++c) gv[c] = sums[g * HIDC + c] / cnt;
    float t[IN_C];
    for (int j = 0; j < IN_C; ++j) {
        float a = b1[j];
        for (int c = 0; c < HIDC; ++c) a += gv[c] * w1[c * IN_C + j];
        t[j] = fmaxf(a, 0.f);
    }
    for (int o = 0; o < OUT_C; ++o) {
        float a = b2[o];
        for (int j = 0; j < IN_C; ++j) a += t[j] * w2[j * OUT_C + o];
        out[g * OUT_C + o] = a;
    }
}

// ---------- host-side helpers ----------
static void launch_edges(const int* ei, int E0, int N, int H, int C,
                         const float* als, const float* ald,
                         const float* htmp, unsigned* maxkey, float* denom,
                         float* acc, hipStream_t stream) {
    const int T = 256;
    int Etot = E0 + N;
    zero_f32_k<<<(N * H + T - 1) / T, T, 0, stream>>>((float*)maxkey, N * H);
    zero_f32_k<<<(N * H + T - 1) / T, T, 0, stream>>>(denom, N * H);
    zero_f32_k<<<(N * H * C + T - 1) / T, T, 0, stream>>>(acc, N * H * C);
    edge_max_k<<<(Etot + T - 1) / T, T, 0, stream>>>(ei, E0, N, H, als, ald, maxkey);
    edge_sum_k<<<(Etot + T - 1) / T, T, 0, stream>>>(ei, E0, N, H, als, ald, maxkey, denom);
    edge_scatter_k<<<(Etot * H + T - 1) / T, T, 0, stream>>>(
        ei, E0, N, H, C, als, ald, maxkey, denom, htmp, acc);
}

static void launch_gemm(const float* X, const float* W, __bf16* Xbf, __bf16* Wtbf,
                        float* OUT, int N, int MC, hipStream_t stream) {
    const int T = 256;
    cvt_bf16_k<<<((N * KDIM) + T - 1) / T, T, 0, stream>>>(X, Xbf, N * KDIM);
    cvt_wt_k<<<((MC * KDIM) + T - 1) / T, T, 0, stream>>>(W, Wtbf, MC);
    dim3 grid((N + 63) / 64);
    dim3 block(32 * (MC / 16));
    size_t sh = (size_t)(MC + 64) * KDIM * sizeof(__bf16);
    gemm_bf16_wmma_k<<<grid, block, sh, stream>>>(Xbf, Wtbf, OUT, N, MC);
}

extern "C" void kernel_launch(void* const* d_in, const int* in_sizes, int n_in,
                              void* d_out, int out_size, void* d_ws, size_t ws_size,
                              hipStream_t stream) {
    const float* x   = (const float*)d_in[0];
    const int*   ei  = (const int*)d_in[1];
    const int*   bat = (const int*)d_in[2];
    const float* W0  = (const float*)d_in[3];
    const float* as0 = (const float*)d_in[4];
    const float* ad0 = (const float*)d_in[5];
    const float* b0  = (const float*)d_in[6];
    const float* W1  = (const float*)d_in[7];
    const float* as1 = (const float*)d_in[8];
    const float* ad1 = (const float*)d_in[9];
    const float* b1  = (const float*)d_in[10];
    const float* W2  = (const float*)d_in[11];
    const float* as2 = (const float*)d_in[12];
    const float* ad2 = (const float*)d_in[13];
    const float* b2  = (const float*)d_in[14];
    const float* l1w = (const float*)d_in[15];
    const float* l1b = (const float*)d_in[16];
    const float* l2w = (const float*)d_in[17];
    const float* l2b = (const float*)d_in[18];
    float* out = (float*)d_out;

    const int N  = in_sizes[0] / IN_C;      // 40000
    const int E0 = in_sizes[1] / 2;         // 640000
    const int G  = out_size / OUT_C;        // 64
    const int HC = HEADS * HIDC;            // 128

    // workspace layout
    float*    ws     = (float*)d_ws;
    float*    htmp   = ws;                                     // N*128
    float*    hacc   = htmp + (size_t)N * HC;                  // N*128
    float*    hfeat  = hacc + (size_t)N * HC;                  // N*128
    float*    als    = hfeat + (size_t)N * HC;                 // N*4
    float*    ald    = als + (size_t)N * HEADS;                // N*4
    unsigned* maxkey = (unsigned*)(ald + (size_t)N * HEADS);   // N*4
    float*    denom  = (float*)(maxkey + (size_t)N * HEADS);   // N*4
    float*    sums   = denom + (size_t)N * HEADS;              // G*32
    float*    cnts   = sums + (size_t)G * HIDC;                // G
    __bf16*   Xbf    = (__bf16*)(cnts + G);                    // N*128 bf16
    __bf16*   Wtbf   = Xbf + (size_t)N * KDIM;                 // 128*128 bf16

    const int T = 256;

    // ---- conv0: x[N,128] -> hfeat[N,128] ----
    launch_gemm(x, W0, Xbf, Wtbf, htmp, N, HC, stream);
    attn_coef_k<<<(N * HEADS + T - 1) / T, T, 0, stream>>>(htmp, as0, ad0, als, ald, N, HEADS, HIDC);
    launch_edges(ei, E0, N, HEADS, HIDC, als, ald, htmp, maxkey, denom, hacc, stream);
    bias_elu_k<<<(N * HC + T - 1) / T, T, 0, stream>>>(hacc, b0, hfeat, N, HC);

    // ---- conv1: hfeat -> hfeat ----
    launch_gemm(hfeat, W1, Xbf, Wtbf, htmp, N, HC, stream);
    attn_coef_k<<<(N * HEADS + T - 1) / T, T, 0, stream>>>(htmp, as1, ad1, als, ald, N, HEADS, HIDC);
    launch_edges(ei, E0, N, HEADS, HIDC, als, ald, htmp, maxkey, denom, hacc, stream);
    bias_elu_k<<<(N * HC + T - 1) / T, T, 0, stream>>>(hacc, b1, hfeat, N, HC);

    // ---- conv2: hfeat[N,128] -> h2[N,32] (1 head, no concat) ----
    launch_gemm(hfeat, W2, Xbf, Wtbf, htmp, N, HIDC, stream);
    attn_coef_k<<<(N + T - 1) / T, T, 0, stream>>>(htmp, as2, ad2, als, ald, N, 1, HIDC);
    launch_edges(ei, E0, N, 1, HIDC, als, ald, htmp, maxkey, denom, hacc, stream);
    bias_elu_k<<<(N * HIDC + T - 1) / T, T, 0, stream>>>(hacc, b2, hfeat, N, HIDC); // h2 in hfeat

    // ---- global mean pool + MLP ----
    zero_f32_k<<<(G * (HIDC + 1) + T - 1) / T, T, 0, stream>>>(sums, G * (HIDC + 1));
    pool_accum_k<<<(N * HIDC + T - 1) / T, T, 0, stream>>>(hfeat, bat, sums, cnts, N);
    mlp_k<<<(G + 63) / 64, 64, 0, stream>>>(sums, cnts, l1w, l1b, l2w, l2b, out, G);
}
